// LSTM_clf_85401129714216
// MI455X (gfx1250) — matrix operand order repaired and checked by reading, not compiled, not run on hardware
//
#include <hip/hip_runtime.h>
#include <math.h>

typedef __attribute__((ext_vector_type(16))) __bf16 v16bf;
typedef __attribute__((ext_vector_type(8)))  float  v8f;
typedef __attribute__((ext_vector_type(4)))  int    v4i;
typedef __attribute__((ext_vector_type(8)))  int    v8i;

union Frag { v16bf v; unsigned u[8]; };

namespace {
constexpr int ZB = 16, ZS = 1024, ZH = 512, ZH2 = 1024, ZG = 2048, ZE = 512, ZL = 10;
}

__device__ __forceinline__ __bf16 f2bf(float f) {
  union { float f; unsigned u; } x; x.f = f;
  unsigned r = (x.u + 0x7FFFu + ((x.u >> 16) & 1u)) >> 16;
  union { unsigned short s; __bf16 h; } y; y.s = (unsigned short)r;
  return y.h;
}

__device__ __forceinline__ float sigm(float x) { return 1.f / (1.f + __expf(-x)); }

// CDNA5 async global->LDS copy (ASYNCcnt-tracked, no VGPR round trip).
__device__ __forceinline__ void async_b128(void* lds, const void* g) {
  const unsigned l = (unsigned)(size_t)lds;   // low 32 bits of flat LDS addr = LDS offset
  asm volatile("global_load_async_to_lds_b128 %0, %1, off"
               :: "v"(l), "v"(g) : "memory");
}
__device__ __forceinline__ void wait_async0() {
  asm volatile("s_wait_asynccnt 0" ::: "memory");
}

// ---------------------------------------------------------------------------
// CDNA5 Tensor Data Mover: 2D bf16 tile (tile_w elems x tile_h rows, row stride
// 'stride' elems) -> LDS at lds_off, padding each 16-DWORD row with 4 DWORDs
// (matches LDT=40 halfword LDS stride). Issued once per wave; TENSORcnt-tracked.
// D# layout per ISA ch.8: group0 = {count, lds_addr, global_addr, type=2},
// group1 = {flags/pad, tensor dims, tile dims, dim0 stride}.
// ---------------------------------------------------------------------------
__device__ __forceinline__ void tdm_load_2d(unsigned lds_off, const void* gaddr,
                                            int tile_w, int tile_h, long long stride) {
  const unsigned long long ga = (unsigned long long)(size_t)gaddr;
  v4i g0;
  g0[0] = 1;                                            // count=1 (valid), user mode
  g0[1] = (int)lds_off;                                 // lds_addr
  g0[2] = (int)(unsigned)ga;                            // global_addr[31:0]
  g0[3] = (int)((unsigned)((ga >> 32) & 0x01FFFFFFull)  // global_addr[56:32]
                | 0x80000000u);                         // type=2 ("image")
  const unsigned td0 = (unsigned)tile_w, td1 = (unsigned)tile_h;
  const unsigned long long st = (unsigned long long)stride;
  v8i g1;
  g1[0] = (int)((1u << 16)       // data_size = 2 bytes
              | (1u << 20)       // pad_enable
              | (3u << 22)       // pad_interval: 16 DWORDs
              | (3u << 25));     // pad_amount: 4 DWORDs
  g1[1] = (int)((td0 & 0xFFFFu) << 16);                       // tensor_dim0[15:0]
  g1[2] = (int)(((td0 >> 16) & 0xFFFFu) | ((td1 & 0xFFFFu) << 16)); // td0 hi | td1 lo
  g1[3] = (int)(((td1 >> 16) & 0xFFFFu) | ((unsigned)tile_w << 16)); // td1 hi | tile_dim0
  g1[4] = (int)(td1 & 0xFFFFu);                               // tile_dim1 (tile_dim2=0)
  g1[5] = (int)(unsigned)st;                                  // tensor_dim0_stride[31:0]
  g1[6] = (int)((st >> 32) & 0xFFFFull);                      // stride[47:32]
  g1[7] = 0;
  asm volatile("tensor_load_to_lds %0, %1" :: "s"(g0), "s"(g1) : "memory");
}

// ---------------------------------------------------------------------------
// Generic bf16 WMMA GEMM: C[b] = A[b](MxK) x B[b] (+bias[n]) (+res).
// BNK=1: B stored (N,K), tiles staged by the Tensor Data Mover (wave 0 only).
// BNK=0: B stored (K,N), transposed into LDS per-lane; A staged async-to-LDS.
// Block 256 threads = 8 waves; 64(M)x128(N) tile; wave tile 32x32 = 4 WMMA/kstep.
// Double-buffered LDS. Requires M%64==0, N%128==0, K%32==0.
// ---------------------------------------------------------------------------
template <int BNK>
__global__ __launch_bounds__(256) void k_gemm(
    const __bf16* __restrict__ A, long long sAb, int lda,
    const __bf16* __restrict__ Bm, long long sBb, int ldb,
    float* __restrict__ C, long long sCb, int ldc,
    const float* __restrict__ bias,
    const float* __restrict__ res, long long sRb, int ldr,
    int M, int N, int K)
{
  constexpr int LDT = 40;                  // padded halfword stride (80B, 16B aligned)
  __shared__ __bf16 At[2][64 * LDT];
  __shared__ __bf16 Bt[2][128 * LDT];
  const int bz = blockIdx.z;
  A  += sAb * bz;
  Bm += sBb * bz;
  C  += sCb * bz;
  const float* resb = res ? res + sRb * bz : nullptr;
  const int m0 = blockIdx.y * 64, n0 = blockIdx.x * 128;
  const int tid = threadIdx.x;
  const int lane = tid & 31, wv = tid >> 5;
  const int l15 = lane & 15, hi = lane >> 4;
  const int aRow = (wv >> 2) * 32;         // 0 or 32
  const int nCol = (wv & 3) * 32;          // 0,32,64,96
  v8f acc00 = {}, acc01 = {}, acc10 = {}, acc11 = {};

  const int rA = tid >> 2, chA = tid & 3;         // A async: 64 rows x 4 16B-chunks
  const int nT = tid & 127, kT = (tid >> 7) * 16; // B(K,N) transpose staging

  auto stage = [&](int buf, int k0) {
    if constexpr (BNK) {
      if (wv == 0) { // one wave drives the DMA engine for both tiles
        tdm_load_2d((unsigned)(size_t)&At[buf][0], A + (long long)m0 * lda + k0,
                    32, 64, lda);
        tdm_load_2d((unsigned)(size_t)&Bt[buf][0], Bm + (long long)n0 * ldb + k0,
                    32, 128, ldb);
      }
    } else {
      async_b128(&At[buf][rA * LDT + chA * 8],
                 A + (long long)(m0 + rA) * lda + k0 + chA * 8);
      #pragma unroll
      for (int i = 0; i < 16; ++i)
        Bt[buf][nT * LDT + kT + i] = Bm[(long long)(k0 + kT + i) * ldb + n0 + nT];
    }
  };
  auto wait_stage = [&]() {
    if constexpr (BNK) {
      if (wv == 0) __builtin_amdgcn_s_wait_tensorcnt(0);
    } else {
      wait_async0();
    }
  };

  stage(0, 0);
  wait_stage();
  __syncthreads();

  int buf = 0;
  for (int k0 = 0; k0 < K; k0 += 32) {
    if (k0 + 32 < K) stage(buf ^ 1, k0 + 32);   // prefetch next tile via DMA

    Frag a0, a1, b0, b1;
    const unsigned* r0 = (const unsigned*)(At[buf] + (aRow + l15) * LDT);
    const unsigned* r1 = (const unsigned*)(At[buf] + (aRow + 16 + l15) * LDT);
    const unsigned* q0 = (const unsigned*)(Bt[buf] + (nCol + l15) * LDT);
    const unsigned* q1 = (const unsigned*)(Bt[buf] + (nCol + 16 + l15) * LDT);
    #pragma unroll
    for (int v = 0; v < 8; ++v) {
      const int dwa = (v < 4 ? v : v + 4) + 4 * hi;   // A pair layout
      const int dwb = (v < 4 ? v : v + 4) + 4 * hi;   // B pair layout
      a0.u[v] = r0[dwa]; a1.u[v] = r1[dwa];
      b0.u[v] = q0[dwb]; b1.u[v] = q1[dwb];
    }
    acc00 = __builtin_amdgcn_wmma_f32_16x16x32_bf16(false, a0.v, false, b0.v, (short)0, acc00, false, false);
    acc01 = __builtin_amdgcn_wmma_f32_16x16x32_bf16(false, a0.v, false, b1.v, (short)0, acc01, false, false);
    acc10 = __builtin_amdgcn_wmma_f32_16x16x32_bf16(false, a1.v, false, b0.v, (short)0, acc10, false, false);
    acc11 = __builtin_amdgcn_wmma_f32_16x16x32_bf16(false, a1.v, false, b1.v, (short)0, acc11, false, false);

    wait_stage();
    __syncthreads();
    buf ^= 1;
  }

  const int nb = n0 + nCol + l15;
  #pragma unroll
  for (int r = 0; r < 8; ++r) {
    const int m = m0 + aRow + r + 8 * hi;
    float v00 = acc00[r], v01 = acc01[r], v10 = acc10[r], v11 = acc11[r];
    if (bias) {
      const float b0v = bias[nb], b1v = bias[nb + 16];
      v00 += b0v; v01 += b1v; v10 += b0v; v11 += b1v;
    }
    if (resb) {
      v00 += resb[(long long)m * ldr + nb];
      v01 += resb[(long long)m * ldr + nb + 16];
      v10 += resb[(long long)(m + 16) * ldr + nb];
      v11 += resb[(long long)(m + 16) * ldr + nb + 16];
    }
    C[(long long)m * ldc + nb] = v00;
    C[(long long)m * ldc + nb + 16] = v01;
    C[(long long)(m + 16) * ldc + nb] = v10;
    C[(long long)(m + 16) * ldc + nb + 16] = v11;
  }
}

// ---------------------------------------------------------------------------
// Persistent bidirectional LSTM recurrence. grid=2 (dir), block=1024 (32 waves).
// Wave w owns hidden units [16w,16w+16) for all 16 batches and all 4 gates:
// cell state c stays in registers; h double-buffered in LDS (bf16) for WMMA A.
// gbuf = precomputed x@W_ih^T + b (S-major: row tt*B+b). whh stored (4H,H) bf16.
// ---------------------------------------------------------------------------
__global__ __launch_bounds__(1024) void k_lstm(
    const float* __restrict__ gF, const float* __restrict__ gR,
    const __bf16* __restrict__ whhF, const __bf16* __restrict__ whhR,
    __bf16* __restrict__ out_bf, float* __restrict__ out_f32, int sb_order)
{
  constexpr int LH = 520;                  // padded halfword stride for h rows
  __shared__ __bf16 hb[2][16 * LH];
  const int dir = blockIdx.x;
  const float* gbuf = dir ? gR : gF;
  const __bf16* whh = dir ? whhR : whhF;
  const int coff = dir * ZH;
  const int tid = threadIdx.x;
  const int lane = tid & 31, wv = tid >> 5;
  const int l15 = lane & 15, hi = lane >> 4;
  const int jb = wv * 16;

  for (int i = tid; i < 16 * LH; i += 1024) hb[0][i] = f2bf(0.f);
  __syncthreads();

  v8f cst = {};
  for (int t = 0; t < ZS; ++t) {
    const int tt = dir ? (ZS - 1 - t) : t;
    const int cur = t & 1, nxt = cur ^ 1;
    v8f acc0 = {}, acc1 = {}, acc2 = {}, acc3 = {};
    #pragma unroll 4
    for (int k0 = 0; k0 < ZH; k0 += 32) {
      Frag af;
      { // A = h (16 batches x 512): row = batch = l15
        const unsigned* ar = (const unsigned*)(&hb[cur][l15 * LH + k0]);
        #pragma unroll
        for (int v = 0; v < 8; ++v) af.u[v] = ar[(v < 4 ? v : v + 4) + 4 * hi];
      }
      Frag b0, b1, b2, b3;
      { // B = W_hh rows n = q*512 + jb + l15, K-contiguous pairs (L2-resident)
        const __bf16* wbase = whh + (long long)(jb + l15) * ZH + k0;
        const unsigned* w0 = (const unsigned*)(wbase);
        const unsigned* w1 = (const unsigned*)(wbase + 512ll * ZH);
        const unsigned* w2 = (const unsigned*)(wbase + 1024ll * ZH);
        const unsigned* w3 = (const unsigned*)(wbase + 1536ll * ZH);
        #pragma unroll
        for (int v = 0; v < 8; ++v) {
          const int dw = (v < 4 ? v : v + 4) + 4 * hi;
          b0.u[v] = w0[dw]; b1.u[v] = w1[dw]; b2.u[v] = w2[dw]; b3.u[v] = w3[dw];
        }
      }
      acc0 = __builtin_amdgcn_wmma_f32_16x16x32_bf16(false, af.v, false, b0.v, (short)0, acc0, false, false);
      acc1 = __builtin_amdgcn_wmma_f32_16x16x32_bf16(false, af.v, false, b1.v, (short)0, acc1, false, false);
      acc2 = __builtin_amdgcn_wmma_f32_16x16x32_bf16(false, af.v, false, b2.v, (short)0, acc2, false, false);
      acc3 = __builtin_amdgcn_wmma_f32_16x16x32_bf16(false, af.v, false, b3.v, (short)0, acc3, false, false);
    }
    // gate math: lane col j = jb+l15, rows b = r + 8*hi (C-layout of 16x16 f32 tile)
    const float* gr = gbuf + (long long)tt * ZB * ZG;
    const int j = jb + l15;
    #pragma unroll
    for (int r = 0; r < 8; ++r) {
      const int b = r + 8 * hi;
      const float* g = gr + (long long)b * ZG + j;
      const float iv = sigm(acc0[r] + g[0]);
      const float fv = sigm(acc1[r] + g[512]);
      const float gv = tanhf(acc2[r] + g[1024]);
      const float ov = sigm(acc3[r] + g[1536]);
      const float c = fv * cst[r] + iv * gv;
      cst[r] = c;
      const float h = ov * tanhf(c);
      hb[nxt][b * LH + j] = f2bf(h);
      const long long orow = sb_order ? ((long long)tt * ZB + b) : ((long long)b * ZS + tt);
      out_bf[orow * ZH2 + coff + j] = f2bf(h);
      if (out_f32) out_f32[orow * ZH2 + coff + j] = h;
    }
    __syncthreads();
  }
}

// ---------------------------------------------------------------------------
// Output projection (N=10) + argmax mask. One wave per (b,s) row.
// ---------------------------------------------------------------------------
__global__ __launch_bounds__(256) void k_outproj(
    const float* __restrict__ x2, const float* __restrict__ ow,
    const float* __restrict__ ob, float* __restrict__ result,
    unsigned char* __restrict__ mask)
{
  const int row = blockIdx.x * 8 + (threadIdx.x >> 5);
  const int lane = threadIdx.x & 31;
  const float* xr = x2 + (long long)row * ZH2;
  float p[ZL];
  #pragma unroll
  for (int l = 0; l < ZL; ++l) p[l] = 0.f;
  for (int k = lane; k < ZH2; k += 32) {
    const float xv = xr[k];
    #pragma unroll
    for (int l = 0; l < ZL; ++l) p[l] = __builtin_fmaf(xv, ow[l * ZH2 + k], p[l]);
  }
  #pragma unroll
  for (int l = 0; l < ZL; ++l)
    for (int s = 16; s > 0; s >>= 1) p[l] += __shfl_xor(p[l], s, 32);
  if (lane == 0) {
    int am = 0; float best = -3.0e38f;
    #pragma unroll
    for (int l = 0; l < ZL; ++l) {
      const float vv = p[l] + ob[l];
      result[(long long)row * ZL + l] = vv;
      if (vv > best) { best = vv; am = l; }
    }
    mask[row] = (unsigned char)(am != 0);
  }
}

// ---------------------------------------------------------------------------
// Two-group masked softmax over one row of scores (in-place) + bf16 copy of P.
// ---------------------------------------------------------------------------
__global__ __launch_bounds__(256) void k_softmax(
    float* __restrict__ sc, __bf16* __restrict__ pbf,
    const unsigned char* __restrict__ mask)
{
  __shared__ float red[256];
  const int tid = threadIdx.x;
  const long long row = blockIdx.x;
  float* x = sc + row * ZS;
  const unsigned char* mk = mask + (row >> 10) * ZS;
  float v[4]; bool m[4];
  float mxE = -3.0e38f, mxO = -3.0e38f;
  #pragma unroll
  for (int i = 0; i < 4; ++i) {
    const int t = tid + 256 * i;
    v[i] = x[t];
    m[i] = mk[t] != 0;
    if (m[i]) mxE = fmaxf(mxE, v[i]); else mxO = fmaxf(mxO, v[i]);
  }
  red[tid] = mxE; __syncthreads();
  for (int s = 128; s > 0; s >>= 1) { if (tid < s) red[tid] = fmaxf(red[tid], red[tid + s]); __syncthreads(); }
  mxE = red[0]; __syncthreads();
  red[tid] = mxO; __syncthreads();
  for (int s = 128; s > 0; s >>= 1) { if (tid < s) red[tid] = fmaxf(red[tid], red[tid + s]); __syncthreads(); }
  mxO = red[0]; __syncthreads();
  float e[4]; float sE = 0.f, sO = 0.f;
  #pragma unroll
  for (int i = 0; i < 4; ++i) {
    e[i] = __expf(v[i] - (m[i] ? mxE : mxO));
    if (m[i]) sE += e[i]; else sO += e[i];
  }
  red[tid] = sE; __syncthreads();
  for (int s = 128; s > 0; s >>= 1) { if (tid < s) red[tid] += red[tid + s]; __syncthreads(); }
  sE = red[0]; __syncthreads();
  red[tid] = sO; __syncthreads();
  for (int s = 128; s > 0; s >>= 1) { if (tid < s) red[tid] += red[tid + s]; __syncthreads(); }
  sO = red[0]; __syncthreads();
  const float rE = sE > 0.f ? 1.f / sE : 0.f;
  const float rO = sO > 0.f ? 1.f / sO : 0.f;
  #pragma unroll
  for (int i = 0; i < 4; ++i) {
    const int t = tid + 256 * i;
    const float pv = e[i] * (m[i] ? rE : rO);
    x[t] = pv;
    pbf[row * ZS + t] = f2bf(pv);
  }
}

// ---------------------------------------------------------------------------
// Embedding gather, (b,s) -> S-major bf16 rows (row = s*B + b) for the GEMM.
// ---------------------------------------------------------------------------
__global__ void k_embed(const int* __restrict__ tok, const float* __restrict__ emb,
                        __bf16* __restrict__ x0) {
  const int blk = blockIdx.x;
  const int b = blk >> 10, s = blk & 1023;
  const long long t = tok[blk];
  const float* e = emb + t * ZE;
  __bf16* o = x0 + ((long long)s * ZB + b) * ZE;
  for (int i = threadIdx.x; i < ZE; i += blockDim.x) o[i] = f2bf(e[i]);
}

__global__ void k_cvt(const float* __restrict__ s, __bf16* __restrict__ d, long long n) {
  const long long i = (long long)blockIdx.x * blockDim.x + threadIdx.x;
  if (i < n) d[i] = f2bf(s[i]);
}

__global__ void k_bias(const float* __restrict__ a, const float* __restrict__ b,
                       float* __restrict__ o) {
  const int i = blockIdx.x * blockDim.x + threadIdx.x;
  o[i] = a[i] + b[i];
}

// ---------------------------------------------------------------------------
extern "C" void kernel_launch(void* const* d_in, const int* in_sizes, int n_in,
                              void* d_out, int out_size, void* d_ws, size_t ws_size,
                              hipStream_t stream) {
  (void)in_sizes; (void)n_in; (void)out_size; (void)ws_size;
  const float* emb  = (const float*)d_in[0];
  const float* wih[4] = {(const float*)d_in[1], (const float*)d_in[5],
                         (const float*)d_in[9], (const float*)d_in[13]};
  const float* whh[4] = {(const float*)d_in[2], (const float*)d_in[6],
                         (const float*)d_in[10], (const float*)d_in[14]};
  const float* bih[4] = {(const float*)d_in[3], (const float*)d_in[7],
                         (const float*)d_in[11], (const float*)d_in[15]};
  const float* bhh[4] = {(const float*)d_in[4], (const float*)d_in[8],
                         (const float*)d_in[12], (const float*)d_in[16]};
  const float* outw = (const float*)d_in[17];
  const float* outb = (const float*)d_in[18];
  const int* tokens = (const int*)d_in[19];

  char* p = (char*)d_ws;
  auto alloc = [&](size_t bytes) -> char* {
    char* r = p; p += (bytes + 255) & ~size_t(255); return r;
  };
  __bf16* x0bf  = (__bf16*)alloc((size_t)ZB * ZS * ZE * 2);
  __bf16* x1bf  = (__bf16*)alloc((size_t)ZB * ZS * ZH2 * 2);
  float*  x2f   = (float*) alloc((size_t)ZB * ZS * ZH2 * 4);
  __bf16* x2bf  = (__bf16*)alloc((size_t)ZB * ZS * ZH2 * 2);
  float*  gF    = (float*) alloc((size_t)ZB * ZS * ZG * 4);
  float*  gR    = (float*) alloc((size_t)ZB * ZS * ZG * 4);
  float*  scores= (float*) alloc((size_t)ZB * ZS * ZS * 4);
  __bf16* pbf   = (__bf16*)alloc((size_t)ZB * ZS * ZS * 2);
  __bf16* wihbf[4]; __bf16* whhbf[4]; float* bsum[4];
  const long long wihN[4] = {(long long)ZG*ZE, (long long)ZG*ZE,
                             (long long)ZG*ZH2, (long long)ZG*ZH2};
  for (int i = 0; i < 4; ++i) wihbf[i] = (__bf16*)alloc((size_t)wihN[i] * 2);
  for (int i = 0; i < 4; ++i) whhbf[i] = (__bf16*)alloc((size_t)ZG * ZH * 2);
  for (int i = 0; i < 4; ++i) bsum[i]  = (float*) alloc((size_t)ZG * 4);
  unsigned char* mask = (unsigned char*)alloc((size_t)ZB * ZS);

  // weight conversions + bias sums
  for (int i = 0; i < 4; ++i) {
    k_cvt<<<dim3((unsigned)((wihN[i] + 255) / 256)), 256, 0, stream>>>(wih[i], wihbf[i], wihN[i]);
    k_cvt<<<dim3((unsigned)(((long long)ZG * ZH + 255) / 256)), 256, 0, stream>>>(whh[i], whhbf[i], (long long)ZG * ZH);
    k_bias<<<dim3(ZG / 256), 256, 0, stream>>>(bih[i], bhh[i], bsum[i]);
  }

  // embedding gather -> bf16 (S-major rows)
  k_embed<<<dim3(ZB * ZS), 128, 0, stream>>>(tokens, emb, x0bf);

  const int Mrows = ZB * ZS;
  // layer 0: input-gate GEMMs (both directions), then persistent recurrence
  k_gemm<1><<<dim3(ZG / 128, Mrows / 64, 1), 256, 0, stream>>>(
      x0bf, 0, ZE, wihbf[0], 0, ZE, gF, 0, ZG, bsum[0], nullptr, 0, 0, Mrows, ZG, ZE);
  k_gemm<1><<<dim3(ZG / 128, Mrows / 64, 1), 256, 0, stream>>>(
      x0bf, 0, ZE, wihbf[1], 0, ZE, gR, 0, ZG, bsum[1], nullptr, 0, 0, Mrows, ZG, ZE);
  k_lstm<<<dim3(2), 1024, 0, stream>>>(gF, gR, whhbf[0], whhbf[1], x1bf, nullptr, 1);

  // layer 1
  k_gemm<1><<<dim3(ZG / 128, Mrows / 64, 1), 256, 0, stream>>>(
      x1bf, 0, ZH2, wihbf[2], 0, ZH2, gF, 0, ZG, bsum[2], nullptr, 0, 0, Mrows, ZG, ZH2);
  k_gemm<1><<<dim3(ZG / 128, Mrows / 64, 1), 256, 0, stream>>>(
      x1bf, 0, ZH2, wihbf[3], 0, ZH2, gR, 0, ZG, bsum[3], nullptr, 0, 0, Mrows, ZG, ZH2);
  k_lstm<<<dim3(2), 1024, 0, stream>>>(gF, gR, whhbf[2], whhbf[3], x2bf, x2f, 0);

  // result + entity mask
  float* outA = (float*)d_out;
  float* outR = outA + (long long)ZB * ZS * ZH2;
  k_outproj<<<dim3(Mrows / 8), 256, 0, stream>>>(x2f, outw, outb, outR, mask);

  // scores = x2 @ x2^T per batch (bf16 WMMA, TDM-staged tiles)
  k_gemm<1><<<dim3(ZS / 128, ZS / 64, ZB), 256, 0, stream>>>(
      x2bf, (long long)ZS * ZH2, ZH2, x2bf, (long long)ZS * ZH2, ZH2,
      scores, (long long)ZS * ZS, ZS, nullptr, nullptr, 0, 0, ZS, ZS, ZH2);

  // two-group masked softmax
  k_softmax<<<dim3(ZB * ZS), 256, 0, stream>>>(scores, pbf, mask);

  // A = P @ x2 + x2 (B is K-major here; residual epilogue)
  k_gemm<0><<<dim3(ZH2 / 128, ZS / 64, ZB), 256, 0, stream>>>(
      pbf, (long long)ZS * ZS, ZS, x2bf, (long long)ZS * ZH2, ZH2,
      outA, (long long)ZS * ZH2, ZH2, nullptr, x2f, (long long)ZS * ZH2, ZH2,
      ZS, ZH2, ZS);
}
